// MulitLatentAttention_73555609911751
// MI455X (gfx1250) — matrix operand order repaired and checked by reading, not compiled, not run on hardware
//
#include <hip/hip_runtime.h>

// ---------------- problem constants ----------------
#define HIDDEN   2048
#define HEADS    16
#define Q_LORA   1024
#define KV_LORA  512
#define NOPE     128
#define ROPE_D   64
#define Q_HEAD   192          // NOPE + ROPE
#define V_HEAD   128
#define BATCH    2
#define SEQ      2048
#define TOKENS   (BATCH * SEQ)

// ---------------- WMMA types ----------------
typedef __attribute__((ext_vector_type(16))) __bf16 v16bf;
typedef __attribute__((ext_vector_type(8)))  float  v8f;

union Frag {
  v16bf v;
  unsigned int   u32[8];
  unsigned short us[16];
};

__device__ __forceinline__ unsigned short f2bf(float f) {
  unsigned int u = __float_as_uint(f);
  u += 0x7FFFu + ((u >> 16) & 1u);          // round-to-nearest-even
  return (unsigned short)(u >> 16);
}

// K index of the low element of 32-bit pair p (0..7) in a 16-bit A fragment
// (contiguous: VGPR0..7 hold K=16*half + 0..15, matching the B fills)
__device__ __forceinline__ int a_pair_k(int p, int half) {
  return 16 * half + 2 * p;
}

// =====================================================================
// Generic GEMM: C[M,N] = A[M,K] * B[K,N], fp32 in/out, bf16 WMMA compute
// block = 256 threads (8 waves); block tile 128 x TNv; K-step 32
// wave tile 32 x (TNv/2):  TNv=128 -> 8 wmma/K-step, TNv=64 -> 4
// =====================================================================
#define TM 128
#define TK 32

template <int TNv>
__global__ __launch_bounds__(256) void gemm_bf16_wmma(
    int M, int N, int K,
    const float* __restrict__ A, int lda,
    const float* __restrict__ B, int ldb,
    float* __restrict__ C, int ldc)
{
  constexpr int NI  = TNv / 32;           // 16-wide wmma n-tiles per wave
  constexpr int BF4 = (TK * TNv) / 1024;  // B float4 per thread
  constexpr int F4R = TNv / 4;            // B float4 per k-row

  __shared__ unsigned short As[TM][TK];    // 8 KB, row-major MxK (bf16 bits)
  __shared__ unsigned short Bs[TNv][TK];   // K-transposed [n][k]

  const int tid  = threadIdx.x;
  const int wave = tid >> 5;
  const int lane = tid & 31;
  const int half = lane >> 4;
  const int l16  = lane & 15;
  const int m0 = blockIdx.y * TM;
  const int n0 = blockIdx.x * TNv;
  const int wm = (wave & 3) * 32;          // wave row offset in tile
  const int wn = (wave >> 2) * (TNv / 2);  // wave col offset in tile

  const v8f vzero = {0.f, 0.f, 0.f, 0.f, 0.f, 0.f, 0.f, 0.f};
  v8f acc[2][NI];
#pragma unroll
  for (int i = 0; i < 2; ++i)
#pragma unroll
    for (int j = 0; j < NI; ++j) acc[i][j] = vzero;

  for (int k0 = 0; k0 < K; k0 += TK) {
    __syncthreads();
    // ---- stage A tile: 128x32 floats -> bf16 (4 float4 per thread) ----
#pragma unroll
    for (int i = 0; i < 4; ++i) {
      const int f4  = tid * 4 + i;        // 0..1023
      const int row = f4 >> 3;            // 8 float4 per 32-wide row
      const int c4  = (f4 & 7) << 2;
      const float* ap = &A[(size_t)(m0 + row) * lda + k0 + c4];
      const float4 v = *reinterpret_cast<const float4*>(ap);
      __builtin_prefetch(ap + TK, 0, 3);  // speculative: next K-slice
      ushort4 us;
      us.x = f2bf(v.x); us.y = f2bf(v.y); us.z = f2bf(v.z); us.w = f2bf(v.w);
      *reinterpret_cast<ushort4*>(&As[row][c4]) = us;
    }
    // ---- stage B tile: 32 x TNv floats -> bf16 transposed ----
#pragma unroll
    for (int i = 0; i < BF4; ++i) {
      const int f4  = tid * BF4 + i;
      const int kr  = f4 / F4R;
      const int nc4 = (f4 % F4R) * 4;
      const float* bp = &B[(size_t)(k0 + kr) * ldb + n0 + nc4];
      const float4 v = *reinterpret_cast<const float4*>(bp);
      __builtin_prefetch(bp + (size_t)TK * ldb, 0, 3);
      Bs[nc4 + 0][kr] = f2bf(v.x);
      Bs[nc4 + 1][kr] = f2bf(v.y);
      Bs[nc4 + 2][kr] = f2bf(v.z);
      Bs[nc4 + 3][kr] = f2bf(v.w);
    }
    __syncthreads();

    Frag a[2], b[NI];
#pragma unroll
    for (int mi = 0; mi < 2; ++mi)
#pragma unroll
      for (int p = 0; p < 8; ++p)
        a[mi].u32[p] = *reinterpret_cast<const unsigned int*>(
            &As[wm + mi * 16 + l16][a_pair_k(p, half)]);
#pragma unroll
    for (int ni = 0; ni < NI; ++ni)
#pragma unroll
      for (int p = 0; p < 8; ++p)
        b[ni].u32[p] = *reinterpret_cast<const unsigned int*>(
            &Bs[wn + ni * 16 + l16][half * 16 + 2 * p]);

#pragma unroll
    for (int mi = 0; mi < 2; ++mi)
#pragma unroll
      for (int ni = 0; ni < NI; ++ni)
        acc[mi][ni] = __builtin_amdgcn_wmma_f32_16x16x32_bf16(
            false, a[mi].v, false, b[ni].v, (short)0, acc[mi][ni], false, false);
  }

#pragma unroll
  for (int mi = 0; mi < 2; ++mi)
#pragma unroll
    for (int ni = 0; ni < NI; ++ni)
#pragma unroll
      for (int r = 0; r < 8; ++r)
        C[(size_t)(m0 + wm + mi * 16 + 8 * half + r) * ldc +
          (n0 + wn + ni * 16 + l16)] = acc[mi][ni][r];
}

// =====================================================================
// RoPE: in-place on q[..., NOPE:NOPE+64] (all heads) and ckv[..., 512:576]
// one thread per (token, unit, pair)
// =====================================================================
__global__ __launch_bounds__(256) void rope_kernel(
    float* __restrict__ q, float* __restrict__ ckv,
    const int* __restrict__ position)
{
  const int idx   = blockIdx.x * blockDim.x + threadIdx.x;
  const int total = TOKENS * (HEADS + 1) * 32;
  if (idx >= total) return;
  const int i    = idx & 31;
  const int rem  = idx >> 5;
  const int unit = rem % (HEADS + 1);
  const int tok  = rem / (HEADS + 1);
  const float pos = (float)position[tok & (SEQ - 1)];
  // inv_freq = theta^(-2i/64); ln(10000) = 9.210340372
  const float ang = pos * __expf(-9.210340371976184f * (float)(2 * i) / 64.0f);
  float s, c;
  __sincosf(ang, &s, &c);
  float* base = (unit < HEADS)
      ? (q   + (size_t)tok * (HEADS * Q_HEAD) + unit * Q_HEAD + NOPE)
      : (ckv + (size_t)tok * (KV_LORA + ROPE_D) + KV_LORA);
  const float t1 = base[i];
  const float t2 = base[i + 32];
  base[i]      = t1 * c - t2 * s;
  base[i + 32] = t2 * c + t1 * s;
}

// =====================================================================
// Flash attention, one (b,h) per blockIdx.y, 128 q-rows per block.
// 8 waves x 16 q-rows; key tiles of 64; bf16 WMMA, f32 accumulate.
// =====================================================================
#define KT 64

__global__ __launch_bounds__(256) void mla_attention(
    const float* __restrict__ qbuf,    // [T][H*192] (rope applied)
    const float* __restrict__ kvbuf,   // [T][H*256] (nope|v)
    const float* __restrict__ ckv,     // [T][576]   (rope applied to [512:576])
    float* __restrict__ obuf)          // [T][H*128]
{
  __shared__ unsigned short Ks[KT][Q_HEAD];    // 24 KB: key x 192 dims
  __shared__ unsigned short Vt[V_HEAD][KT];    // 16 KB: v-dim x key (transposed)
  __shared__ unsigned short Ps[8][16][KT];     // 16 KB: per-wave P scratch

  const int tid  = threadIdx.x;
  const int wave = tid >> 5;
  const int lane = tid & 31;
  const int half = lane >> 4;
  const int l16  = lane & 15;
  const int b    = blockIdx.y >> 4;
  const int h    = blockIdx.y & 15;
  const int q0   = blockIdx.x * 128;
  const size_t tokBase = (size_t)b * SEQ;
  const float scale = 0.07216878364870323f;    // 1/sqrt(192)

  // ---- load this wave's 16x192 Q as 6 A-fragments (pre-scaled) ----
  Frag qf[6];
  {
    const int qrow = q0 + wave * 16 + l16;     // A layout: M = lane%16
    const float* qp = qbuf + (tokBase + qrow) * (size_t)(HEADS * Q_HEAD) + h * Q_HEAD;
#pragma unroll
    for (int f = 0; f < 6; ++f)
#pragma unroll
      for (int p = 0; p < 8; ++p) {
        const int kk = f * 32 + a_pair_k(p, half);
        const float2 xv = *reinterpret_cast<const float2*>(&qp[kk]);
        qf[f].us[2 * p]     = f2bf(xv.x * scale);
        qf[f].us[2 * p + 1] = f2bf(xv.y * scale);
      }
  }

  const v8f vzero = {0.f, 0.f, 0.f, 0.f, 0.f, 0.f, 0.f, 0.f};
  v8f o[8];
#pragma unroll
  for (int t = 0; t < 8; ++t) o[t] = vzero;
  float rowM[8], rowL[8];
#pragma unroll
  for (int r = 0; r < 8; ++r) { rowM[r] = -3.0e38f; rowL[r] = 0.0f; }

  for (int j0 = 0; j0 < SEQ; j0 += KT) {
    __syncthreads();
    // ---- K tile: 64 keys x 192 dims; V tile transposed Vt[v-dim][key] ----
    // 4 threads per key row: 48 K-dims + 32 V-dims each (no divisions)
    {
      const int row  = tid >> 2;            // 0..63
      const int part = tid & 3;
      const size_t tok = tokBase + j0 + row;
      const float* kp = kvbuf + tok * (size_t)(HEADS * (NOPE + V_HEAD)) + h * (NOPE + V_HEAD);
      const float* rp = ckv + tok * (size_t)(KV_LORA + ROPE_D) + KV_LORA;
      const int d0 = part * 48;
#pragma unroll
      for (int i = 0; i < 48; ++i) {
        const int d = d0 + i;
        Ks[row][d] = f2bf((d < NOPE) ? kp[d] : rp[d - NOPE]);
      }
      const int vd0 = part * 32;
#pragma unroll
      for (int i = 0; i < 32; ++i)
        Vt[vd0 + i][row] = f2bf(kp[NOPE + vd0 + i]);
    }
    __syncthreads();

    // ---- scores: 16 q-rows x 64 keys, contraction over 192 dims ----
    v8f sc[4];
#pragma unroll
    for (int nb = 0; nb < 4; ++nb) sc[nb] = vzero;
#pragma unroll
    for (int f = 0; f < 6; ++f) {
#pragma unroll
      for (int nb = 0; nb < 4; ++nb) {
        Frag bf;
#pragma unroll
        for (int p = 0; p < 8; ++p)
          bf.u32[p] = *reinterpret_cast<const unsigned int*>(
              &Ks[nb * 16 + l16][f * 32 + half * 16 + 2 * p]);
        sc[nb] = __builtin_amdgcn_wmma_f32_16x16x32_bf16(
            false, qf[f].v, false, bf.v, (short)0, sc[nb], false, false);
      }
    }

    // ---- online softmax (rows r+8*half; reduce across 16-lane groups) ----
    float nm[8], sfac[8];
#pragma unroll
    for (int r = 0; r < 8; ++r) {
      float v = fmaxf(fmaxf(sc[0][r], sc[1][r]), fmaxf(sc[2][r], sc[3][r]));
      v = fmaxf(v, __shfl_xor(v, 1, 32));
      v = fmaxf(v, __shfl_xor(v, 2, 32));
      v = fmaxf(v, __shfl_xor(v, 4, 32));
      v = fmaxf(v, __shfl_xor(v, 8, 32));
      nm[r]   = fmaxf(rowM[r], v);
      sfac[r] = __expf(rowM[r] - nm[r]);
    }
#pragma unroll
    for (int nb = 0; nb < 4; ++nb)
#pragma unroll
      for (int r = 0; r < 8; ++r)
        sc[nb][r] = __expf(sc[nb][r] - nm[r]);
#pragma unroll
    for (int r = 0; r < 8; ++r) {
      float s = sc[0][r] + sc[1][r] + sc[2][r] + sc[3][r];
      s += __shfl_xor(s, 1, 32);
      s += __shfl_xor(s, 2, 32);
      s += __shfl_xor(s, 4, 32);
      s += __shfl_xor(s, 8, 32);
      rowL[r] = rowL[r] * sfac[r] + s;
      rowM[r] = nm[r];
    }
#pragma unroll
    for (int t = 0; t < 8; ++t)
#pragma unroll
      for (int r = 0; r < 8; ++r) o[t][r] *= sfac[r];

    // ---- P (C layout) -> per-wave LDS row-major 16x64 bf16 ----
#pragma unroll
    for (int nb = 0; nb < 4; ++nb)
#pragma unroll
      for (int r = 0; r < 8; ++r)
        Ps[wave][8 * half + r][nb * 16 + l16] = f2bf(sc[nb][r]);

    // ---- O += P @ V (contraction over 64 keys, N = 128 v-dims) ----
#pragma unroll
    for (int kc = 0; kc < 2; ++kc) {
      Frag pa;
#pragma unroll
      for (int p = 0; p < 8; ++p)
        pa.u32[p] = *reinterpret_cast<const unsigned int*>(
            &Ps[wave][l16][kc * 32 + a_pair_k(p, half)]);
#pragma unroll
      for (int nt = 0; nt < 8; ++nt) {
        Frag vb;
#pragma unroll
        for (int p = 0; p < 8; ++p)
          vb.u32[p] = *reinterpret_cast<const unsigned int*>(
              &Vt[nt * 16 + l16][kc * 32 + half * 16 + 2 * p]);
        o[nt] = __builtin_amdgcn_wmma_f32_16x16x32_bf16(
            false, pa.v, false, vb.v, (short)0, o[nt], false, false);
      }
    }
  }

  // ---- normalize and store ----
  float rinv[8];
#pragma unroll
  for (int r = 0; r < 8; ++r) rinv[r] = 1.0f / rowL[r];
  const size_t rowBase = tokBase + q0 + wave * 16 + 8 * half;
#pragma unroll
  for (int nt = 0; nt < 8; ++nt)
#pragma unroll
    for (int r = 0; r < 8; ++r)
      obuf[(rowBase + r) * (size_t)HIDDEN + h * V_HEAD + nt * 16 + l16] =
          o[nt][r] * rinv[r];
}

// =====================================================================
// host launcher
// =====================================================================
extern "C" void kernel_launch(void* const* d_in, const int* in_sizes, int n_in,
                              void* d_out, int out_size, void* d_ws, size_t ws_size,
                              hipStream_t stream)
{
  (void)in_sizes; (void)n_in; (void)out_size; (void)ws_size;
  const float* x        = (const float*)d_in[0];
  const int*   position = (const int*)d_in[1];
  const float* Wq_down  = (const float*)d_in[2];
  const float* Wq_up    = (const float*)d_in[3];
  const float* Wkv_down = (const float*)d_in[4];
  const float* Wkv_up   = (const float*)d_in[5];
  const float* Wout     = (const float*)d_in[6];
  float* out = (float*)d_out;

  char* ws = (char*)d_ws;
  size_t off = 0;
  float* q_lora = (float*)(ws + off); off += (size_t)TOKENS * Q_LORA * sizeof(float);
  float* qbuf   = (float*)(ws + off); off += (size_t)TOKENS * HEADS * Q_HEAD * sizeof(float);
  float* ckvb   = (float*)(ws + off); off += (size_t)TOKENS * (KV_LORA + ROPE_D) * sizeof(float);
  float* kvbuf  = (float*)(ws + off); off += (size_t)TOKENS * HEADS * (NOPE + V_HEAD) * sizeof(float);
  float* attn   = (float*)(ws + off); off += (size_t)TOKENS * HIDDEN * sizeof(float);

  const dim3 blk(256);
  // q_lora = x @ Wq_down
  gemm_bf16_wmma<128><<<dim3(Q_LORA / 128, TOKENS / TM), blk, 0, stream>>>(
      TOKENS, Q_LORA, HIDDEN, x, HIDDEN, Wq_down, Q_LORA, q_lora, Q_LORA);
  // q = q_lora @ Wq_up
  gemm_bf16_wmma<128><<<dim3((HEADS * Q_HEAD) / 128, TOKENS / TM), blk, 0, stream>>>(
      TOKENS, HEADS * Q_HEAD, Q_LORA, q_lora, Q_LORA, Wq_up, HEADS * Q_HEAD,
      qbuf, HEADS * Q_HEAD);
  // ckv = x @ Wkv_down   (N=576 -> 64-wide tiles)
  gemm_bf16_wmma<64><<<dim3((KV_LORA + ROPE_D) / 64, TOKENS / TM), blk, 0, stream>>>(
      TOKENS, KV_LORA + ROPE_D, HIDDEN, x, HIDDEN, Wkv_down, KV_LORA + ROPE_D,
      ckvb, KV_LORA + ROPE_D);
  // kv = ckv[:, :512] @ Wkv_up   (A row stride 576)
  gemm_bf16_wmma<128><<<dim3((HEADS * (NOPE + V_HEAD)) / 128, TOKENS / TM), blk, 0, stream>>>(
      TOKENS, HEADS * (NOPE + V_HEAD), KV_LORA, ckvb, KV_LORA + ROPE_D,
      Wkv_up, HEADS * (NOPE + V_HEAD), kvbuf, HEADS * (NOPE + V_HEAD));
  // rope in-place on q_rope and k_rope
  rope_kernel<<<dim3((TOKENS * (HEADS + 1) * 32) / 256), blk, 0, stream>>>(
      qbuf, ckvb, position);
  // attention
  mla_attention<<<dim3(SEQ / 128, BATCH * HEADS), blk, 0, stream>>>(
      qbuf, kvbuf, ckvb, attn);
  // out = attn @ Wout
  gemm_bf16_wmma<128><<<dim3(HIDDEN / 128, TOKENS / TM), blk, 0, stream>>>(
      TOKENS, HIDDEN, HIDDEN, attn, HIDDEN, Wout, HIDDEN, out, HIDDEN);
}
